// EGNNLayer_64802466562191
// MI455X (gfx1250) — matrix operand order repaired and checked by reading, not compiled, not run on hardware
//
#include <hip/hip_runtime.h>

// ---------------------------------------------------------------------------
// E(3)-equivariant GNN layer for MI455X (gfx1250), wave32 + WMMA f16 + TDM.
//
// Edge kernel (1 wave = 16 edges), two fused GEMM stages:
//   stage 1:  tp = [F_s @ W12^ | F_v[i] @ W34^[i]]  (768-K WMMA, scaled)
//   stage 2:  messages = tp @ Wm + bm               (160x160 WMMA)
//   scatter:  atomic-add into agg[row]
// Node gather uses the Tensor Data Mover in GATHER mode (16 x 16-bit row
// indices per descriptor == one 16-edge tile), tracked with TENSORcnt.
// Node kernel: out = x + concat(x, agg) @ Wu + bu   (WMMA)
// Unrolling is capped at 2 to stay under 256 VGPRs (no s_set_vgpr_msb
// churn) and keep 4+ waves/SIMD for latency hiding.
// ---------------------------------------------------------------------------

typedef __attribute__((ext_vector_type(16))) _Float16     v16h;
typedef __attribute__((ext_vector_type(8)))  _Float16     v8h;
typedef __attribute__((ext_vector_type(8)))  float        v8f;
typedef __attribute__((ext_vector_type(4)))  float        v4f;
typedef __attribute__((ext_vector_type(4)))  unsigned int v4u;
typedef __attribute__((ext_vector_type(8)))  int          v8i;
typedef __attribute__((ext_vector_type(4)))  int          v4i;

#define N_NODES   50000
#define N_EDGES   800000
#define D_NODE    160
#define D_EDGE    32
#define D_HID     160
#define NW        4          // waves per block (128 threads)

#define INV_SQRT2 0.7071067811865476f
#define A1 0.04419417382415922f   // 1/sqrt(64*8)
#define A2 0.03608439182435161f   // 1/(sqrt(3)*sqrt(32*8))
#define A3 0.04419417382415922f   // 1/sqrt(64*8)
#define A4 0.0625f                // 1/sqrt(32*8)

// fragment counts (each fragment = 512 f16 = [lane][16 halves])
#define NFRAG_WS (24 * 4)         // W12^: K=768, N=64
#define NFRAG_WV (3 * 24 * 2)     // W34^: 3 x (K=768, N=32)
#define NFRAG_WM (5 * 10)         // Wm:   K=160, N=160
#define NFRAG_WU (10 * 10)        // Wu:   K=320, N=160

// ---------------------------------------------------------------- zero agg
__global__ void zero_agg_kernel(float* __restrict__ agg, int n) {
    for (int i = blockIdx.x * blockDim.x + threadIdx.x; i < n;
         i += gridDim.x * blockDim.x)
        agg[i] = 0.0f;
}

// ---- scaled W1/W2 -> B fragments (B 32x16: lanes 0-15 K=h<8?h:h+8, N=lane; 16-31 K=8+(h<8?h:h+8))
__global__ void repack_wS(const float* __restrict__ W1, const float* __restrict__ W2,
                          _Float16* __restrict__ out) {
    int t = blockIdx.x * blockDim.x + threadIdx.x;
    if (t >= NFRAG_WS * 512) return;
    int h = t & 15, lane = (t >> 4) & 31, frag = t >> 9;
    int nt = frag & 3, kt = frag >> 2;
    int k = kt * 32 + ((lane < 16) ? 0 : 8) + h + ((h < 8) ? 0 : 8);
    int c = nt * 16 + (lane & 15);                 // 0..63
    float v;
    if (k < 512) { int a = k >> 3, b = k & 7;
        v = A1 * INV_SQRT2 * W1[(a * 8 + b) * 64 + c];         // W1[64][8][64]
    } else { int j = k - 512; int a = j >> 3, b = j & 7;
        v = A2 * INV_SQRT2 * W2[(a * 8 + b) * 64 + c];         // W2[32][8][64]
    }
    out[t] = (_Float16)v;
}

__global__ void repack_wV(const float* __restrict__ W3, const float* __restrict__ W4,
                          _Float16* __restrict__ out) {
    int t = blockIdx.x * blockDim.x + threadIdx.x;
    if (t >= NFRAG_WV * 512) return;
    int h = t & 15, lane = (t >> 4) & 31, frag = t >> 9;
    int nt = frag % 2, kt = (frag / 2) % 24;       // i = frag / 48 (same W for all i)
    int k = kt * 32 + ((lane < 16) ? 0 : 8) + h + ((h < 8) ? 0 : 8);
    int c = nt * 16 + (lane & 15);                 // 0..31
    float v;
    if (k < 512) { int a = k >> 3, b = k & 7;
        v = A3 * INV_SQRT2 * W3[(a * 8 + b) * 32 + c];         // W3[64][8][32]
    } else { int j = k - 512; int a = j >> 3, b = j & 7;
        v = A4 * INV_SQRT2 * W4[(a * 8 + b) * 32 + c];         // W4[32][8][32]
    }
    out[t] = (_Float16)v;
}

__global__ void repack_wM(const float* __restrict__ Wm, _Float16* __restrict__ out) {
    int t = blockIdx.x * blockDim.x + threadIdx.x;
    if (t >= NFRAG_WM * 512) return;
    int h = t & 15, lane = (t >> 4) & 31, frag = t >> 9;
    int nt = frag % 10, kt = frag / 10;
    int k = kt * 32 + ((lane < 16) ? 0 : 8) + h + ((h < 8) ? 0 : 8);  // tp channel (matches reference order)
    int n = nt * 16 + (lane & 15);
    out[t] = (_Float16)Wm[k * D_HID + n];
}

__global__ void repack_wU(const float* __restrict__ Wu, _Float16* __restrict__ out) {
    int t = blockIdx.x * blockDim.x + threadIdx.x;
    if (t >= NFRAG_WU * 512) return;
    int h = t & 15, lane = (t >> 4) & 31, frag = t >> 9;
    int nt = frag % 10, kt = frag / 10;
    int k = kt * 32 + ((lane < 16) ? 0 : 8) + h + ((h < 8) ? 0 : 8);
    int n = nt * 16 + (lane & 15);
    out[t] = (_Float16)Wu[k * D_NODE + n];
}

// --------------------------------------------------------------- edge kernel
__global__ __launch_bounds__(NW * 32) void edge_kernel(
        const float* __restrict__ xfeat, const int* __restrict__ eidx,
        const float* __restrict__ eattr, const _Float16* __restrict__ wS,
        const _Float16* __restrict__ wV, const _Float16* __restrict__ wM,
        const float* __restrict__ bm, float* __restrict__ agg) {
    __shared__ float    s_x [NW][16][D_NODE];   // 40 KB: TDM gather dest; aliased by tp later
    __shared__ _Float16 s_ea[NW][16][D_EDGE];   // 4 KB

    const int w    = threadIdx.x >> 5;
    const int lane = threadIdx.x & 31;
    const int tile = blockIdx.x * NW + w;       // grid exact: 800000/16/NW
    const int e0   = tile * 16;
    const int* colIdx = eidx + N_EDGES;         // edge_index[1] = source nodes

    // ---- TDM gather of 16 node-feature rows (640 B each) into s_x[w] ------
    {
        unsigned lds  = (unsigned)(uintptr_t)&s_x[w][0][0];
        unsigned long long ga = (unsigned long long)(uintptr_t)xfeat;
        v4u g0;
        g0[0] = 0x80000001u;                    // count=1, gather_mode=1, 16-bit idx
        g0[1] = lds;                            // lds_addr
        g0[2] = (unsigned)(ga & 0xFFFFFFFFu);   // global_addr[31:0]
        g0[3] = (unsigned)((ga >> 32) & 0x01FFFFFFu) | 0x80000000u; // addr[56:32], type=2
        v8i g1;
        g1[0] = 0x00020000;                     // data_size=2 (4B), wg_mask=0
        g1[1] = (int)((D_NODE & 0xFFFF) << 16); // tensor_dim0[15:0]=160
        g1[2] = (int)(((unsigned)N_NODES & 0xFFFFu) << 16); // tensor_dim1[15:0]
        g1[3] = (int)((D_NODE & 0xFFFF) << 16); // tile_dim0=160
        g1[4] = 16;                             // tile_dim1 = #valid indices
        g1[5] = D_NODE;                         // tensor_dim0_stride=160
        g1[6] = 0; g1[7] = 0;
        v4i g2, g3;                             // 16 x 16-bit row indices
#pragma unroll
        for (int j = 0; j < 4; ++j) {
            unsigned lo = (unsigned)colIdx[e0 + 2 * j]     & 0xFFFFu;
            unsigned hi = (unsigned)colIdx[e0 + 2 * j + 1] & 0xFFFFu;
            g2[j] = (int)(lo | (hi << 16));
        }
#pragma unroll
        for (int j = 0; j < 4; ++j) {
            unsigned lo = (unsigned)colIdx[e0 + 8 + 2 * j]     & 0xFFFFu;
            unsigned hi = (unsigned)colIdx[e0 + 8 + 2 * j + 1] & 0xFFFFu;
            g3[j] = (int)(lo | (hi << 16));
        }
        v8i gz = {0, 0, 0, 0, 0, 0, 0, 0};      // unused trailing group (6-arg form)
        __builtin_amdgcn_tensor_load_to_lds(g0, g1, g2, g3, gz, 0);
    }

    // ---- stage edge attrs (f16) while the TDM runs ------------------------
    for (int idx = lane; idx < 16 * D_EDGE; idx += 32) {
        int mm = idx >> 5, c = idx & 31;
        s_ea[w][mm][c] = (_Float16)eattr[(e0 + mm) * D_EDGE + c];
    }
    __builtin_amdgcn_s_wait_tensorcnt(0);
    __syncthreads();

    const int m     = lane & 15;
    const int koff  = (lane < 16) ? 0 : 8;      // A 16x32 f16 lane layout
    const int qbase = (lane < 16) ? 0 : 1;      // (koff + h-term) >> 3 contribution
    const int mrow  = (lane < 16) ? 0 : 8;      // C lane layout

    // per-lane register copy of this edge-row's attrs: es = ea[0..7], ev = ea[8+b*3+i]
    _Float16 ea[32];
#pragma unroll
    for (int c = 0; c < 32; ++c) ea[c] = s_ea[w][m][c];

    // ---- stage 1: tensor-product GEMMs -> tp tile -------------------------
    // For fixed (kt, lane):  a-index = kt*4 + qbase (+2 for h>=8),  b-index = h&7.
    v8f acc_s[4] = {};
    v8f acc_v[3][2] = {};
#pragma unroll
    for (int f = 0; f < 4; ++f) {
        // region 1: features k<512  (kt = 0..15): outer(xs, es/ev_i)
#pragma unroll 2
        for (int kt = 0; kt < 16; ++kt) {
            _Float16 x0 = (_Float16)s_x[w][m][kt * 4 + qbase];
            _Float16 x1 = (_Float16)s_x[w][m][kt * 4 + qbase + 2];
            v16h a;
#pragma unroll
            for (int h = 0; h < 8; ++h) {
                _Float16 e = (f == 0) ? ea[h] : ea[8 + h * 3 + (f - 1)];
                a[h]     = x0 * e;
                a[h + 8] = x1 * e;
            }
            if (f == 0) {
                const _Float16* wp = wS + ((kt * 4) << 9) + (lane << 4);
#pragma unroll
                for (int nt = 0; nt < 4; ++nt) {
                    v16h b = *(const v16h*)(wp + (nt << 9));
                    acc_s[nt] = __builtin_amdgcn_wmma_f32_16x16x32_f16(
                        false, a, false, b, (short)0, acc_s[nt], false, false);
                }
            } else {
                const _Float16* wp = wV + ((((f - 1) * 24 + kt) * 2) << 9) + (lane << 4);
#pragma unroll
                for (int nt = 0; nt < 2; ++nt) {
                    v16h b = *(const v16h*)(wp + (nt << 9));
                    acc_v[f - 1][nt] = __builtin_amdgcn_wmma_f32_16x16x32_f16(
                        false, a, false, b, (short)0, acc_v[f - 1][nt], false, false);
                }
            }
        }
        // region 2: features k>=512 (kt = 16..23): dot3 / outer(xv_i, es)
#pragma unroll 2
        for (int kt2 = 0; kt2 < 8; ++kt2) {
            int kt = kt2 + 16;
            int a0 = kt2 * 4 + qbase;            // xv row (0..31)
            v16h a;
            if (f == 0) {                        // dot3(xv_a, ev_b)
                const float* xp0 = &s_x[w][m][64 + a0 * 3];
                const float* xp1 = &s_x[w][m][64 + (a0 + 2) * 3];
                float x00 = xp0[0], x01 = xp0[1], x02 = xp0[2];
                float x10 = xp1[0], x11 = xp1[1], x12 = xp1[2];
#pragma unroll
                for (int h = 0; h < 8; ++h) {
                    float e0 = (float)ea[8 + h * 3 + 0];
                    float e1 = (float)ea[8 + h * 3 + 1];
                    float e2 = (float)ea[8 + h * 3 + 2];
                    a[h]     = (_Float16)(x00 * e0 + x01 * e1 + x02 * e2);
                    a[h + 8] = (_Float16)(x10 * e0 + x11 * e1 + x12 * e2);
                }
            } else {                             // outer(xv_i, es)
                int i = f - 1;
                _Float16 x0 = (_Float16)s_x[w][m][64 + a0 * 3 + i];
                _Float16 x1 = (_Float16)s_x[w][m][64 + (a0 + 2) * 3 + i];
#pragma unroll
                for (int h = 0; h < 8; ++h) {
                    a[h]     = x0 * ea[h];
                    a[h + 8] = x1 * ea[h];
                }
            }
            if (f == 0) {
                const _Float16* wp = wS + ((kt * 4) << 9) + (lane << 4);
#pragma unroll
                for (int nt = 0; nt < 4; ++nt) {
                    v16h b = *(const v16h*)(wp + (nt << 9));
                    acc_s[nt] = __builtin_amdgcn_wmma_f32_16x16x32_f16(
                        false, a, false, b, (short)0, acc_s[nt], false, false);
                }
            } else {
                const _Float16* wp = wV + ((((f - 1) * 24 + kt) * 2) << 9) + (lane << 4);
#pragma unroll
                for (int nt = 0; nt < 2; ++nt) {
                    v16h b = *(const v16h*)(wp + (nt << 9));
                    acc_v[f - 1][nt] = __builtin_amdgcn_wmma_f32_16x16x32_f16(
                        false, a, false, b, (short)0, acc_v[f - 1][nt], false, false);
                }
            }
        }
    }

    // ---- C-fragment -> A-fragment relayout of tp through LDS --------------
    __syncthreads();                                  // s_x reads done; safe to alias
    _Float16* s_tp = reinterpret_cast<_Float16*>(&s_x[w][0][0]);  // [16][160] f16
#pragma unroll
    for (int nt = 0; nt < 4; ++nt) {
        int c = nt * 16 + (lane & 15);                // scalar channel 0..63
#pragma unroll
        for (int r = 0; r < 8; ++r)
            s_tp[(r + mrow) * D_HID + c] = (_Float16)acc_s[nt][r];
    }
#pragma unroll
    for (int i = 0; i < 3; ++i)
#pragma unroll
        for (int nt = 0; nt < 2; ++nt) {
            int c = nt * 16 + (lane & 15);            // vector channel 0..31
#pragma unroll
            for (int r = 0; r < 8; ++r)
                s_tp[(r + mrow) * D_HID + 64 + 3 * c + i] = (_Float16)acc_v[i][nt][r];
        }
    __syncthreads();

    // ---- stage 2: messages = tp @ Wm (vectorized 16-B LDS A loads) --------
    v8f accm[10] = {};
#pragma unroll 2
    for (int kt = 0; kt < 5; ++kt) {
        const _Float16* ap = s_tp + m * D_HID + kt * 32 + koff;
        v8h lo = *(const v8h*)(ap);
        v8h hi = *(const v8h*)(ap + 16);
        v16h a;
#pragma unroll
        for (int h = 0; h < 8; ++h) { a[h] = lo[h]; a[h + 8] = hi[h]; }
        const _Float16* wp = wM + ((kt * 10) << 9) + (lane << 4);
#pragma unroll
        for (int nt = 0; nt < 10; ++nt) {
            v16h b = *(const v16h*)(wp + (nt << 9));
            accm[nt] = __builtin_amdgcn_wmma_f32_16x16x32_f16(
                false, a, false, b, (short)0, accm[nt], false, false);
        }
    }

    // ---- + bm, scatter-add into agg ---------------------------------------
    int rows[8];
#pragma unroll
    for (int r = 0; r < 8; ++r) rows[r] = eidx[e0 + r + mrow];   // edge_index[0]
#pragma unroll
    for (int nt = 0; nt < 10; ++nt) {
        int n = nt * 16 + (lane & 15);
        float bn = bm[n];
#pragma unroll
        for (int r = 0; r < 8; ++r)
            unsafeAtomicAdd(&agg[rows[r] * D_HID + n], accm[nt][r] + bn);
    }
}

// --------------------------------------------------------------- node kernel
__global__ __launch_bounds__(NW * 32) void node_kernel(
        const float* __restrict__ xfeat, const float* __restrict__ agg,
        const _Float16* __restrict__ wU, const float* __restrict__ bu,
        float* __restrict__ out) {
    const int w    = threadIdx.x >> 5;
    const int lane = threadIdx.x & 31;
    const int tile = blockIdx.x * NW + w;
    if (tile >= N_NODES / 16) return;           // wave-uniform guard
    const int n0   = tile * 16;
    const int m    = lane & 15;
    const int koff = (lane < 16) ? 0 : 8;
    const int node = n0 + m;

    v8f acc[10] = {};
#pragma unroll 2
    for (int kt = 0; kt < 10; ++kt) {
        const float* src = (kt < 5)
            ? (xfeat + node * D_NODE + kt * 32 + koff)
            : (agg   + node * D_HID  + (kt - 5) * 32 + koff);
        v4f p0 = *(const v4f*)(src);
        v4f p1 = *(const v4f*)(src + 4);
        v4f p2 = *(const v4f*)(src + 16);
        v4f p3 = *(const v4f*)(src + 20);
        v16h a;
#pragma unroll
        for (int h = 0; h < 4; ++h) {
            a[h]      = (_Float16)p0[h];
            a[h + 4]  = (_Float16)p1[h];
            a[h + 8]  = (_Float16)p2[h];
            a[h + 12] = (_Float16)p3[h];
        }
        const _Float16* wp = wU + ((kt * 10) << 9) + (lane << 4);
#pragma unroll
        for (int nt = 0; nt < 10; ++nt) {
            v16h b = *(const v16h*)(wp + (nt << 9));
            acc[nt] = __builtin_amdgcn_wmma_f32_16x16x32_f16(
                false, a, false, b, (short)0, acc[nt], false, false);
        }
    }

    const int mrow = (lane < 16) ? 0 : 8;
#pragma unroll
    for (int nt = 0; nt < 10; ++nt) {
        int n = nt * 16 + (lane & 15);
        float bn = bu[n];
#pragma unroll
        for (int r = 0; r < 8; ++r) {
            int nd = n0 + r + mrow;
            out[nd * D_NODE + n] = xfeat[nd * D_NODE + n] + acc[nt][r] + bn;
        }
    }
}

// ---------------------------------------------------------------------------
extern "C" void kernel_launch(void* const* d_in, const int* in_sizes, int n_in,
                              void* d_out, int out_size, void* d_ws, size_t ws_size,
                              hipStream_t stream) {
    const float* xfeat = (const float*)d_in[0];   // (50000, 160)
    const int*   eidx  = (const int*)d_in[1];     // (2, 800000)
    const float* eattr = (const float*)d_in[2];   // (800000, 32)
    // d_in[3] node_attr_scalar_raw: unused by reference
    const float* W1 = (const float*)d_in[4];
    const float* W2 = (const float*)d_in[5];
    const float* W3 = (const float*)d_in[6];
    const float* W4 = (const float*)d_in[7];
    const float* Wm = (const float*)d_in[8];
    const float* bm = (const float*)d_in[9];
    const float* Wu = (const float*)d_in[10];
    const float* bu = (const float*)d_in[11];

    float*    agg = (float*)d_ws;                        // 50000*160 f32 = 32 MB
    _Float16* pS  = (_Float16*)(agg + N_NODES * D_HID);
    _Float16* pV  = pS + NFRAG_WS * 512;
    _Float16* pM  = pV + NFRAG_WV * 512;
    _Float16* pU  = pM + NFRAG_WM * 512;

    zero_agg_kernel<<<4096, 256, 0, stream>>>(agg, N_NODES * D_HID);
    repack_wS<<<(NFRAG_WS * 512) / 256, 256, 0, stream>>>(W1, W2, pS);
    repack_wV<<<(NFRAG_WV * 512) / 256, 256, 0, stream>>>(W3, W4, pV);
    repack_wM<<<(NFRAG_WM * 512) / 256, 256, 0, stream>>>(Wm, pM);
    repack_wU<<<(NFRAG_WU * 512) / 256, 256, 0, stream>>>(Wu, pU);
    edge_kernel<<<N_EDGES / 16 / NW, NW * 32, 0, stream>>>(
        xfeat, eidx, eattr, pS, pV, pM, bm, agg);
    node_kernel<<<(N_NODES / 16 + NW - 1) / NW, NW * 32, 0, stream>>>(
        xfeat, agg, pU, bu, (float*)d_out);
}
